// SelfAttention_81561428951084
// MI455X (gfx1250) — matrix operand order repaired and checked by reading, not compiled, not run on hardware
//
#include <hip/hip_runtime.h>

typedef __bf16 v16bf __attribute__((ext_vector_type(16)));
typedef float  v8f   __attribute__((ext_vector_type(8)));

#define EMBED 1024
#define SLEN  1024
#define NB    2
#define NG    64
#define HD    16
#define MROWS (NB * SLEN)   // 2048 flattened token rows

__device__ __forceinline__ unsigned short f2bf(float f) {
  union { float f; unsigned u; } cv; cv.f = f;
  unsigned r = cv.u + 0x7FFFu + ((cv.u >> 16) & 1u);   // round-to-nearest-even
  return (unsigned short)(r >> 16);
}

// ---------------------------------------------------------------- cast fp32 -> bf16
__global__ void cast_f32_bf16(const float* __restrict__ src,
                              unsigned short* __restrict__ dst, int n) {
  int i = blockIdx.x * blockDim.x + threadIdx.x;
  int stride = gridDim.x * blockDim.x;
  for (; i < n; i += stride) dst[i] = f2bf(src[i]);
}

// cast + transpose a 1024x1024 weight: dst[n*1024+k] = bf16(src[k*1024+n])
__global__ void cast_transpose_bf16(const float* __restrict__ src,
                                    unsigned short* __restrict__ dst) {
  int i = blockIdx.x * blockDim.x + threadIdx.x;
  int stride = gridDim.x * blockDim.x;
  for (; i < 1024 * 1024; i += stride) {
    int k = i >> 10, n = i & 1023;
    dst[(size_t)n * 1024 + k] = f2bf(src[i]);
  }
}

// ---------------------------------------------------------------- bf16 WMMA GEMM
// C(2048x1024) = A(2048x1024) * B(1024x1024); B supplied TRANSPOSED (BT[n][k]).
// MODE 0: z in {0,1,2} selects Wq/Wk/Wv.
//   z=0,1 -> bf16 into [b,g,s,h] layout (Q,K);  z=2 -> bf16 into [b,g,h,s] (V^T).
// MODE 1: fp32 C + bias to Cout.
template <int MODE>
__global__ void __launch_bounds__(256)
gemm_bf16_wmma(const unsigned short* __restrict__ A,
               const unsigned short* __restrict__ BTbase,
               unsigned short* __restrict__ Cqkv,
               const float* __restrict__ bias,
               float* __restrict__ Cout) {
  const int tid  = threadIdx.x;
  const int lane = tid & 31;
  const int w    = tid >> 5;
  const int l15  = lane & 15;
  const int hi   = lane >> 4;          // 0: lanes 0-15, 1: lanes 16-31
  const int m0   = blockIdx.y * 128;
  const int n0   = blockIdx.x * 128;
  const int z    = blockIdx.z;
  const unsigned short* BT = BTbase + (size_t)z * (1024 * 1024);

  __shared__ unsigned short ldsA[128 * 40];   // [m][k], padded stride 40
  __shared__ unsigned short ldsBt[128 * 40];  // [n][k], padded stride 40

  const int wm = w >> 2;   // 0..1 : 64-row band
  const int wn = w & 3;    // 0..3 : 32-col band

  v8f acc[4][2];
  for (int i = 0; i < 4; i++)
    for (int j = 0; j < 2; j++)
      for (int e = 0; e < 8; e++) acc[i][j][e] = 0.0f;

  for (int k0 = 0; k0 < 1024; k0 += 32) {
    __syncthreads();
    // A tile 128x32 and BT tile 128x32: both plain row-major vec8 copies
    for (int v = tid; v < 512; v += 256) {
      int row = v >> 2;
      int col = (v & 3) * 8;
      uint4 da = *reinterpret_cast<const uint4*>(A + (size_t)(m0 + row) * 1024 + k0 + col);
      *reinterpret_cast<uint4*>(&ldsA[row * 40 + col]) = da;
      uint4 db = *reinterpret_cast<const uint4*>(BT + (size_t)(n0 + row) * 1024 + k0 + col);
      *reinterpret_cast<uint4*>(&ldsBt[row * 40 + col]) = db;
      if (k0 + 32 < 1024) {  // prefetch next K-slab (global_prefetch_b8)
        __builtin_prefetch(A + (size_t)(m0 + row) * 1024 + k0 + 32 + col, 0, 1);
        __builtin_prefetch(BT + (size_t)(n0 + row) * 1024 + k0 + 32 + col, 0, 1);
      }
    }
    __syncthreads();

    // Fragments per ISA 16-bit layouts.
    v16bf afr[4], bfr[2];
    for (int ms = 0; ms < 4; ms++) {
      int m = wm * 64 + ms * 16 + l15;
      int koff = hi * 8;  // A: lanes<16 -> K 0..7 & 16..23 ; lanes>=16 -> K 8..15 & 24..31
      union { uint4 u[2]; v16bf v; } ua;
      ua.u[0] = *reinterpret_cast<const uint4*>(&ldsA[m * 40 + koff]);
      ua.u[1] = *reinterpret_cast<const uint4*>(&ldsA[m * 40 + 16 + koff]);
      afr[ms] = ua.v;
    }
    for (int ns = 0; ns < 2; ns++) {
      int n = wn * 32 + ns * 16 + l15;
      int koff = hi * 8;  // B: lanes<16 -> K 0..7 & 16..23 ; lanes>=16 -> K 8..15 & 24..31
      union { uint4 u[2]; v16bf v; } ub;
      ub.u[0] = *reinterpret_cast<const uint4*>(&ldsBt[n * 40 + koff]);
      ub.u[1] = *reinterpret_cast<const uint4*>(&ldsBt[n * 40 + 16 + koff]);
      bfr[ns] = ub.v;
    }
    for (int ms = 0; ms < 4; ms++)
      for (int ns = 0; ns < 2; ns++)
        acc[ms][ns] = __builtin_amdgcn_wmma_f32_16x16x32_bf16(
            false, afr[ms], false, bfr[ns], (short)0, acc[ms][ns], false, false);
  }

  // Epilogue. C layout: VGPR r -> M = hi*8+r, N = l15.
  for (int ms = 0; ms < 4; ms++)
    for (int ns = 0; ns < 2; ns++) {
      int gm0 = m0 + wm * 64 + ms * 16 + hi * 8;     // 8 consecutive rows
      int gn  = n0 + wn * 32 + ns * 16 + l15;
      if (MODE == 0) {
        int bb = gm0 >> 10;
        int g = gn >> 4, h = gn & 15;
        if (z == 2) {
          // V^T layout [b,g,h,s]: 8 consecutive s -> one 128-bit store
          union { unsigned short s[8]; uint4 u; } pk;
          for (int r = 0; r < 8; r++) pk.s[r] = f2bf(acc[ms][ns][r]);
          int s0 = gm0 & 1023;
          *reinterpret_cast<uint4*>(
              Cqkv + (size_t)2 * (MROWS * 1024) +
              (((size_t)(bb * NG + g) * HD + h) * SLEN + s0)) = pk.u;
        } else {
          for (int r = 0; r < 8; r++) {
            int s = (gm0 + r) & 1023;
            Cqkv[(size_t)z * (MROWS * 1024) +
                 (((size_t)(bb * NG + g) * SLEN + s) * HD + h)] = f2bf(acc[ms][ns][r]);
          }
        }
      } else {
        for (int r = 0; r < 8; r++)
          Cout[(size_t)(gm0 + r) * 1024 + gn] = acc[ms][ns][r] + bias[gn];
      }
    }
}

// ---------------------------------------------------------------- attention
// One block = (b, g, 64-query tile). Full score rows live in LDS (64x1028 fp32,
// ~257KB of the 320KB WGP LDS). bf16 WMMA for QK^T and att*V, fp32 softmax.
__global__ void __launch_bounds__(256)
attn_kernel(const unsigned short* __restrict__ Qb,
            const unsigned short* __restrict__ Kb,
            const unsigned short* __restrict__ Vt,   // [b,g,h,s]
            unsigned short* __restrict__ Yb) {
  extern __shared__ char smem[];
  const int ASTR = 1028;
  float* att           = (float*)smem;                             // 64 x 1028 fp32
  unsigned short* qtil = (unsigned short*)(smem + 64 * ASTR * 4);  // 64 x 16 bf16
  float* pbuf          = (float*)(smem + 64 * ASTR * 4 + 2048);    // 8 x 16 x 16 fp32

  const int tid = threadIdx.x, lane = tid & 31, w = tid >> 5;
  const int l15 = lane & 15, hi = lane >> 4;
  const int q0 = blockIdx.x * 64;
  const int g  = blockIdx.y;
  const int b  = blockIdx.z;
  const size_t bg = (size_t)(b * NG + g) * SLEN;  // row base into Q/K (units of HD)
  const int Lp = q0 + 64;                         // padded causal key extent

  // stage Q tile (64x16 bf16)
  {
    const unsigned* src = reinterpret_cast<const unsigned*>(Qb + (bg + q0) * HD);
    unsigned* dst = reinterpret_cast<unsigned*>(qtil);
    for (int i = tid; i < 512; i += 256) dst[i] = src[i];
  }
  __syncthreads();

  // ---- scores: S = Q K^T * 0.25, causal mask, written fp32 to LDS
  {
    const int qsub = w >> 1, kph = w & 1;
    union { uint4 u[2]; v16bf v; } ua;
    int m = qsub * 16 + l15;
    // head dim 16: valid K<16 only. lanes<16 carry K0..15, lanes>=16 carry zero padding.
    ua.u[0] = hi ? make_uint4(0, 0, 0, 0) : *reinterpret_cast<const uint4*>(&qtil[m * 16]);
    ua.u[1] = hi ? make_uint4(0, 0, 0, 0) : *reinterpret_cast<const uint4*>(&qtil[m * 16 + 8]);
    v16bf aq = ua.v;

    for (int key0 = kph * 16; key0 < Lp; key0 += 32) {
      union { uint4 u[2]; v16bf v; } ub;
      if (hi == 0) {  // lanes 0-15: B K-rows 0..15 = head dims (full K row, 32B)
        const uint4* kp = reinterpret_cast<const uint4*>(Kb + (bg + key0 + l15) * HD);
        ub.u[0] = kp[0];
        ub.u[1] = kp[1];
      } else {        // lanes 16-31: B K-rows 16..31 = zero padding
        ub.u[0] = make_uint4(0, 0, 0, 0);
        ub.u[1] = make_uint4(0, 0, 0, 0);
      }
      v8f c;
      for (int e = 0; e < 8; e++) c[e] = 0.0f;
      c = __builtin_amdgcn_wmma_f32_16x16x32_bf16(false, aq, false, ub.v,
                                                  (short)0, c, false, false);
      for (int r = 0; r < 8; r++) {
        int rl = hi * 8 + r;
        int qg = q0 + qsub * 16 + rl;
        int kg = key0 + l15;
        float sv = (kg <= qg) ? c[r] * 0.25f : -__builtin_inff();
        att[(qsub * 16 + rl) * ASTR + kg] = sv;
      }
    }
  }
  __syncthreads();

  // ---- exact softmax per row (wave w owns rows w*8 .. w*8+7)
  for (int i = 0; i < 8; i++) {
    int row = w * 8 + i;
    float* ar = att + row * ASTR;
    int L = q0 + row + 1;  // causal valid length
    float mx = -__builtin_inff();
    for (int j = lane; j < L; j += 32) mx = fmaxf(mx, ar[j]);
    for (int o = 16; o > 0; o >>= 1) mx = fmaxf(mx, __shfl_xor(mx, o, 32));
    float sum = 0.0f;
    for (int j = lane; j < Lp; j += 32) {
      float e = __expf(ar[j] - mx);  // masked entries: exp(-inf) -> 0
      ar[j] = e;
      sum += e;
    }
    for (int o = 16; o > 0; o >>= 1) sum += __shfl_xor(sum, o, 32);
    float inv = 1.0f / sum;
    for (int j = lane; j < Lp; j += 32) ar[j] *= inv;
  }
  __syncthreads();

  // ---- y = att * V, split-K over 2 halves x 4 query sub-tiles (8 waves)
  {
    const int qsub = w & 3, hlf = w >> 2;
    v8f acc;
    for (int e = 0; e < 8; e++) acc[e] = 0.0f;
    const unsigned short* vrow = Vt + ((size_t)(b * NG + g) * HD + l15) * SLEN;
    for (int c0 = hlf * 32; c0 < Lp; c0 += 64) {
      // A frag from att (fp32 -> bf16), vectorized LDS reads
      union { unsigned short s[16]; v16bf v; } ua;
      int row = qsub * 16 + l15;
      const float* ar = att + row * ASTR + c0;
      int ko = hi * 8;
      float4 f0 = *reinterpret_cast<const float4*>(ar + ko);
      float4 f1 = *reinterpret_cast<const float4*>(ar + ko + 4);
      float4 f2 = *reinterpret_cast<const float4*>(ar + 16 + ko);
      float4 f3 = *reinterpret_cast<const float4*>(ar + 16 + ko + 4);
      ua.s[0] = f2bf(f0.x);  ua.s[1] = f2bf(f0.y);  ua.s[2] = f2bf(f0.z);  ua.s[3] = f2bf(f0.w);
      ua.s[4] = f2bf(f1.x);  ua.s[5] = f2bf(f1.y);  ua.s[6] = f2bf(f1.z);  ua.s[7] = f2bf(f1.w);
      ua.s[8] = f2bf(f2.x);  ua.s[9] = f2bf(f2.y);  ua.s[10] = f2bf(f2.z); ua.s[11] = f2bf(f2.w);
      ua.s[12] = f2bf(f3.x); ua.s[13] = f2bf(f3.y); ua.s[14] = f2bf(f3.z); ua.s[15] = f2bf(f3.w);
      // B frag from V^T: lane = column h, s values hi*8..hi*8+7 and 16+hi*8..16+hi*8+7
      union { uint4 u[2]; v16bf v; } ub;
      const uint4* vp = reinterpret_cast<const uint4*>(vrow + c0 + hi * 8);
      ub.u[0] = vp[0];
      ub.u[1] = vp[2];
      acc = __builtin_amdgcn_wmma_f32_16x16x32_bf16(false, ua.v, false, ub.v,
                                                    (short)0, acc, false, false);
    }
    for (int r = 0; r < 8; r++)
      pbuf[w * 256 + (hi * 8 + r) * 16 + l15] = acc[r];
  }
  __syncthreads();

  // combine split-K halves, write y as bf16 into [b,s,e] layout
  for (int idx = tid; idx < 1024; idx += 256) {
    int row = idx >> 4, h = idx & 15;
    int qsub = row >> 4, rl = row & 15;
    float v0 = pbuf[qsub * 256 + rl * 16 + h];
    float v1 = pbuf[(4 + qsub) * 256 + rl * 16 + h];
    Yb[((size_t)(b * SLEN) + q0 + row) * 1024 + (size_t)g * 16 + h] = f2bf(v0 + v1);
  }
}

// ---------------------------------------------------------------- launch
extern "C" void kernel_launch(void* const* d_in, const int* in_sizes, int n_in,
                              void* d_out, int out_size, void* d_ws, size_t ws_size,
                              hipStream_t stream) {
  (void)in_sizes; (void)n_in; (void)out_size; (void)ws_size;
  const float* x  = (const float*)d_in[0];
  const float* Wq = (const float*)d_in[1];
  const float* Wk = (const float*)d_in[2];
  const float* Wv = (const float*)d_in[3];
  const float* Wo = (const float*)d_in[4];
  const float* bo = (const float*)d_in[5];
  float* out = (float*)d_out;

  // workspace carve (bf16 halves stored as ushort)
  unsigned short* xb   = (unsigned short*)d_ws;        // 2048x1024
  unsigned short* wbT  = xb + 2 * 1024 * 1024;         // WqT|WkT|WvT, 3x [n][k]
  unsigned short* woT  = wbT + 3 * 1024 * 1024;        // WoT, [n][k]
  unsigned short* qkv  = woT + 1024 * 1024;            // Q|K in [b,g,s,h]; V in [b,g,h,s]
  unsigned short* yb   = qkv + (size_t)3 * 2 * 1024 * 1024;  // 2048x1024

  cast_f32_bf16<<<256, 256, 0, stream>>>(x, xb, 2 * 1024 * 1024);
  cast_transpose_bf16<<<256, 256, 0, stream>>>(Wq, wbT);
  cast_transpose_bf16<<<256, 256, 0, stream>>>(Wk, wbT + 1024 * 1024);
  cast_transpose_bf16<<<256, 256, 0, stream>>>(Wv, wbT + 2 * 1024 * 1024);
  cast_transpose_bf16<<<256, 256, 0, stream>>>(Wo, woT);

  // fused QKV projection (z selects weight + destination/layout)
  gemm_bf16_wmma<0><<<dim3(8, 16, 3), 256, 0, stream>>>(xb, wbT, qkv, nullptr, nullptr);

  // attention: 257KB fp32 score tile + Q tile + split-K partials in LDS
  size_t smem = (size_t)64 * 1028 * 4 + 2048 + 8192;  // 273,408 B < 320KB WGP LDS
  hipFuncSetAttribute((const void*)attn_kernel,
                      hipFuncAttributeMaxDynamicSharedMemorySize, (int)smem);
  attn_kernel<<<dim3(16, 64, 2), 256, smem, stream>>>(
      qkv, qkv + 2 * 1024 * 1024, qkv + 4 * 1024 * 1024, yb);

  // output projection + bias (fp32 out)
  gemm_bf16_wmma<1><<<dim3(8, 16, 1), 256, 0, stream>>>(yb, woT, nullptr, bo, out);
}
